// DGCNNSeg_3899830304781
// MI455X (gfx1250) — matrix operand order repaired and checked by reading, not compiled, not run on hardware
//
#include <hip/hip_runtime.h>

#define DEV __device__ __forceinline__

typedef __bf16 bf16;
typedef __attribute__((ext_vector_type(16))) bf16 v16bf;
typedef __attribute__((ext_vector_type(8)))  bf16 v8bf;
typedef __attribute__((ext_vector_type(8)))  float v8f;

static constexpr int BB   = 4;
static constexpr int NN   = 4096;
static constexpr int KNBR = 20;
static constexpr int NCLS = 13;

DEV float lrelu(float h) { return h >= 0.f ? h : 0.2f * h; }

DEV v8f wmma_bf16(v16bf a, v16bf b, v8f c) {
  // codegen-confirmed: v_wmma_f32_16x16x32_bf16 (f32 accumulate)
  return __builtin_amdgcn_wmma_f32_16x16x32_bf16(false, a, false, b, (short)0, c,
                                                 false, false);
}

// A fragment (16x32 bf16, MxK) from f32 row. Per ISA layout: lane m / m+16
// hold row m, K chunks {kb..kb+7} and {kb+16..kb+23}, kb = kstep*32 + khalf*8.
DEV v16bf afrag_f32(const float* __restrict__ row, int kb) {
  v16bf a;
#pragma unroll
  for (int i = 0; i < 8; ++i) {
    a[i]     = (bf16)row[kb + i];
    a[8 + i] = (bf16)row[kb + 16 + i];
  }
  return a;
}

// Same fragment from a bf16 row in LDS: two aligned 16B chunk loads.
DEV v16bf afrag_bf16(const bf16* row, int kb) {
  v8bf lo = *(const v8bf*)(row + kb);
  v8bf hi = *(const v8bf*)(row + kb + 16);
  return __builtin_shufflevector(lo, hi, 0, 1, 2, 3, 4, 5, 6, 7,
                                 8, 9, 10, 11, 12, 13, 14, 15);
}

// B fragment (32x16 bf16, KxN) from pre-packed bf16 weights. Lane n holds
// column n; K chunks {k0..k0+7} and {k0+16..k0+23}, k0 = kbase + 8*khalf: two aligned 16B loads.
DEV v16bf bfrag_packed(const bf16* __restrict__ wrow, int k0) {
  v8bf lo = *(const v8bf*)(wrow + k0);
  v8bf hi = *(const v8bf*)(wrow + k0 + 16);
  return __builtin_shufflevector(lo, hi, 0, 1, 2, 3, 4, 5, 6, 7,
                                 8, 9, 10, 11, 12, 13, 14, 15);
}

// ---------------------------------------------------------------------------
// Pack f32 weights (Co x ldw, column window [w_off, w_off+Ci)) into bf16 with
// K padded with zeros to KPAD (multiple of 32). Runs once per launch, ~1 MB.
// ---------------------------------------------------------------------------
__launch_bounds__(256) __global__
void pack_w_kernel(const float* __restrict__ W, int ldw, int w_off, int Ci,
                   int KPAD, int Co, bf16* __restrict__ out) {
  int e = blockIdx.x * 256 + threadIdx.x;
  if (e >= Co * KPAD) return;
  int o = e / KPAD, c = e % KPAD;
  out[e] = (bf16)(c < Ci ? W[(size_t)o * ldw + w_off + c] : 0.f);
}

// ---------------------------------------------------------------------------
// kNN: one thread per point, fused inner-product + running top-20 (largest pd)
// ---------------------------------------------------------------------------
template <int C>
__launch_bounds__(256) __global__
void knn_kernel(const float* __restrict__ pts, int ld, int off,
                int* __restrict__ idx_out) {
  int gid = blockIdx.x * blockDim.x + threadIdx.x;
  if (gid >= BB * NN) return;
  int b = gid / NN, n = gid % NN;
  const float* base = pts + (size_t)b * NN * ld + off;

  float self[C];
  float sq = 0.f;
#pragma unroll
  for (int c = 0; c < C; ++c) {
    self[c] = base[(size_t)n * ld + c];
    sq += self[c] * self[c];
  }

  float bd[KNBR];
  int   bi[KNBR];
#pragma unroll
  for (int j = 0; j < KNBR; ++j) { bd[j] = -3.0e38f; bi[j] = 0; }
  float minv = -3.0e38f;
  int   minp = 0;

  for (int m = 0; m < NN; ++m) {
    const float* rm = base + (size_t)m * ld;
    float dot = 0.f, sqm = 0.f;
#pragma unroll
    for (int c = 0; c < C; ++c) {
      float f = rm[c];
      dot += self[c] * f;
      sqm += f * f;
    }
    float pd = 2.f * dot - sq - sqm;
    if (pd > minv) {  // rare path: replace current min and rescan
      bd[minp] = pd;
      bi[minp] = m;
      minv = bd[0]; minp = 0;
#pragma unroll
      for (int j = 1; j < KNBR; ++j)
        if (bd[j] < minv) { minv = bd[j]; minp = j; }
    }
  }
  int* op = idx_out + (size_t)gid * KNBR;
#pragma unroll
  for (int j = 0; j < KNBR; ++j) op[j] = bi[j];
}

// ---------------------------------------------------------------------------
// Fused edge-conv stage: stage edge features (nb - c | c) in LDS as bf16,
// conv_a (-> 64ch, lrelu), optional conv_b (64->64, lrelu), max over the 20
// edges per point. Block = 16 points * 20 edges = 320 rows = 20 WMMA M-tiles
// distributed over 8 waves. Weights are pre-packed bf16 (KPAD columns).
// ---------------------------------------------------------------------------
template <int CIN, bool TWO>
__launch_bounds__(256) __global__
void edge_conv_kernel(const float* __restrict__ pts, int ld_pts, int pts_off,
                      const int* __restrict__ knn,
                      const bf16* __restrict__ Wa, const float* __restrict__ sa,
                      const float* __restrict__ ba,
                      const bf16* __restrict__ Wb, const float* __restrict__ sb,
                      const float* __restrict__ bb,
                      float* __restrict__ out, int ld_out, int out_off) {
  constexpr int EC     = 2 * CIN;
  constexpr int KPAD   = (EC + 31) & ~31;   // 32 (stage1) or 128 (stage2/3)
  constexpr int KSTEPS = KPAD / 32;
  constexpr int ROWS   = 16 * KNBR;         // 320

  __shared__ bf16 sfeat[ROWS][KPAD];         // gathered edge features
  __shared__ bf16 sh2[ROWS][64];             // final per-edge activations
  __shared__ bf16 sh1[TWO ? 8 : 1][16][64];  // per-wave conv_a output tile
  __shared__ int  sj[ROWS];                  // neighbor indices

  int blocksPerB = NN / 16;
  int b    = blockIdx.x / blocksPerB;
  int n0   = (blockIdx.x % blocksPerB) * 16;
  int wave = threadIdx.x >> 5;
  int lane = threadIdx.x & 31;    // wave32
  int l15  = lane & 15;
  int khalf = lane >> 4;

  const float* pbase = pts + (size_t)b * NN * ld_pts + pts_off;

  // --- gather pass: neighbor ids, then branchless edge-feature build ---
  for (int r = threadIdx.x; r < ROWS; r += 256) {
    int p = r / KNBR, kk = r % KNBR;
    sj[r] = knn[((size_t)b * NN + n0 + p) * KNBR + kk];
  }
  __syncthreads();
  for (int e = threadIdx.x; e < ROWS * KPAD; e += 256) {
    int r = e / KPAD, c = e % KPAD;      // KPAD is a power of two -> shifts
    int p = r / KNBR;
    const float* pn = pbase + (size_t)(n0 + p) * ld_pts;
    const float* pj = pbase + (size_t)sj[r] * ld_pts;
    int mc = (c < CIN) ? c : (c - CIN);  // clamped in-row index
    mc = (mc < CIN) ? mc : 0;
    const float* ap = (c < CIN) ? pj : pn;
    float v = ap[mc] - ((c < CIN) ? pn[mc] : 0.f);
    v = (c < EC) ? v : 0.f;              // zero padding (stage1 only)
    sfeat[r][c] = (bf16)v;
  }
  __syncthreads();

  // --- per-wave WMMA tiles ---
  for (int t = wave; t < KNBR; t += 8) {
    const bf16* arow = &sfeat[t * 16 + l15][0];
    v8f acc[4] = {{0,0,0,0,0,0,0,0}, {0,0,0,0,0,0,0,0},
                  {0,0,0,0,0,0,0,0}, {0,0,0,0,0,0,0,0}};
#pragma unroll
    for (int ks = 0; ks < KSTEPS; ++ks) {
      v16bf a = afrag_bf16(arow, ks * 32 + khalf * 8);
      int k0 = ks * 32 + khalf * 8;
#pragma unroll
      for (int ot = 0; ot < 4; ++ot) {
        v16bf bf = bfrag_packed(Wa + (size_t)(ot * 16 + l15) * KPAD, k0);
        acc[ot] = wmma_bf16(a, bf, acc[ot]);
      }
    }
    // conv_a epilogue; C layout: vgpr jr -> row jr (+8 for hi lane half)
#pragma unroll
    for (int ot = 0; ot < 4; ++ot) {
      int col = ot * 16 + l15;
      float s = sa[col], bia = ba[col];
#pragma unroll
      for (int jr = 0; jr < 8; ++jr) {
        int m = jr + (khalf ? 8 : 0);
        float h = lrelu(acc[ot][jr] * s + bia);
        if (TWO) sh1[wave][m][col]    = (bf16)h;
        else     sh2[t * 16 + m][col] = (bf16)h;
      }
    }
    if (TWO) {
      asm volatile("s_wait_dscnt 0" ::: "memory");  // wave-local LDS RAW
      v8f acc2[4] = {{0,0,0,0,0,0,0,0}, {0,0,0,0,0,0,0,0},
                     {0,0,0,0,0,0,0,0}, {0,0,0,0,0,0,0,0}};
#pragma unroll
      for (int ks = 0; ks < 2; ++ks) {
        v16bf a = afrag_bf16(&sh1[wave][l15][0], ks * 32 + khalf * 8);
        int k0 = ks * 32 + khalf * 8;
#pragma unroll
        for (int ot = 0; ot < 4; ++ot) {
          v16bf bf = bfrag_packed(Wb + (size_t)(ot * 16 + l15) * 64, k0);
          acc2[ot] = wmma_bf16(a, bf, acc2[ot]);
        }
      }
#pragma unroll
      for (int ot = 0; ot < 4; ++ot) {
        int col = ot * 16 + l15;
        float s = sb[col], bia = bb[col];
#pragma unroll
        for (int jr = 0; jr < 8; ++jr) {
          int m = jr + (khalf ? 8 : 0);
          sh2[t * 16 + m][col] = (bf16)lrelu(acc2[ot][jr] * s + bia);
        }
      }
    }
  }
  __syncthreads();
  // max over the 20 edges of each point, write into cat buffer column slice
  for (int e = threadIdx.x; e < 16 * 64; e += 256) {
    int p = e >> 6, ch = e & 63;
    float best = -3.0e38f;
#pragma unroll
    for (int kk = 0; kk < KNBR; ++kk) {
      float v = (float)sh2[p * KNBR + kk][ch];
      best = v > best ? v : best;
    }
    out[((size_t)b * NN + n0 + p) * ld_out + out_off + ch] = best;
  }
}

// ---------------------------------------------------------------------------
// Generic pointwise-conv GEMM with packed bf16 weights (Co x Kdim, Kdim % 32
// == 0): out[r, co] = lrelu((A[r,:]@W[co,:] + extra)*s + b).
// Block = 8 waves; wave computes a 16x16 tile over the full K loop.
// ---------------------------------------------------------------------------
__launch_bounds__(256) __global__
void gemm_kernel(const float* __restrict__ in, int ld_in,
                 const bf16* __restrict__ Wp, int Kdim,
                 const float* __restrict__ sc, const float* __restrict__ bias,
                 const float* __restrict__ extra, int Co,
                 float* __restrict__ out, int ld_out) {
  int wave = threadIdx.x >> 5;
  int lane = threadIdx.x & 31;
  int l15 = lane & 15, khalf = lane >> 4;
  int r0  = blockIdx.x * 128 + wave * 16;
  int c0  = blockIdx.y * 16;
  int col = c0 + l15;

  const float* arow = in + (size_t)(r0 + l15) * ld_in;
  const bf16*  wrow = Wp + (size_t)col * Kdim;

  v8f acc = {0, 0, 0, 0, 0, 0, 0, 0};
  for (int k = 0; k < Kdim; k += 32) {
    v16bf a  = afrag_f32(arow, k + khalf * 8);
    v16bf bf = bfrag_packed(wrow, k + khalf * 8);
    acc = wmma_bf16(a, bf, acc);
  }

  float s = sc[col], bi = bias[col];
  float ex = 0.f;
  if (extra) {
    int bidx = r0 / NN;  // 128-row blocks never straddle a batch (4096 % 128 == 0)
    ex = extra[(size_t)bidx * Co + col];
  }
#pragma unroll
  for (int jr = 0; jr < 8; ++jr) {
    int row = r0 + jr + (khalf ? 8 : 0);
    out[(size_t)row * ld_out + col] = lrelu((acc[jr] + ex) * s + bi);
  }
}

// g[b,o] = max_n h6[b,n,o]  (coalesced over o)
__launch_bounds__(256) __global__
void max_over_n_kernel(const float* __restrict__ h6, float* __restrict__ g) {
  int b = blockIdx.x;
  int o = blockIdx.y * 256 + threadIdx.x;
  const float* base = h6 + (size_t)b * NN * 1024 + o;
  float best = -3.0e38f;
  for (int n = 0; n < NN; ++n) {
    float v = base[(size_t)n * 1024];
    best = v > best ? v : best;
  }
  g[b * 1024 + o] = best;
}

// gw[b,o] = sum_{c<1024} w7[o, c] * g[b, c]   (w7 is (512, 1216) f32)
__launch_bounds__(256) __global__
void gw_kernel(const float* __restrict__ w7, const float* __restrict__ g,
               float* __restrict__ gw) {
  int b = blockIdx.y;
  int o = blockIdx.x * 256 + threadIdx.x;
  const float* wrow = w7 + (size_t)o * 1216;
  const float* gb   = g + b * 1024;
  float s = 0.f;
  for (int c = 0; c < 1024; ++c) s += wrow[c] * gb[c];
  gw[b * 512 + o] = s;
}

// out[b, o, n] = b9[o] + w9[o,:] @ h8[b, n, :]
__launch_bounds__(256) __global__
void head_kernel(const float* __restrict__ h8, const float* __restrict__ w9,
                 const float* __restrict__ b9, float* __restrict__ outp) {
  int gid = blockIdx.x * 256 + threadIdx.x;
  if (gid >= BB * NN * NCLS) return;
  int o = gid % NCLS;
  int n = (gid / NCLS) % NN;
  int b = gid / (NCLS * NN);
  const float* row  = h8 + ((size_t)b * NN + n) * 256;
  const float* wrow = w9 + o * 256;
  float s = b9[o];
  for (int c = 0; c < 256; ++c) s += wrow[c] * row[c];
  outp[((size_t)b * NCLS + o) * NN + n] = s;
}

// ---------------------------------------------------------------------------
extern "C" void kernel_launch(void* const* d_in, const int* in_sizes, int n_in,
                              void* d_out, int out_size, void* d_ws, size_t ws_size,
                              hipStream_t stream) {
  (void)in_sizes; (void)n_in; (void)out_size; (void)ws_size;

  const float* x = (const float*)d_in[0];
  const float* w[9]; const float* sc[9]; const float* bi[9];
  for (int j = 1; j <= 8; ++j) {
    w[j]  = (const float*)d_in[1 + 3 * (j - 1)];
    sc[j] = (const float*)d_in[2 + 3 * (j - 1)];
    bi[j] = (const float*)d_in[3 + 3 * (j - 1)];
  }
  const float* w9 = (const float*)d_in[25];
  const float* b9 = (const float*)d_in[26];
  float* outp = (float*)d_out;

  // ---- workspace layout ----
  // packed bf16 weights first (aligned), then f32/int buffers. Total ~82 MB.
  bf16* wp = (bf16*)d_ws;
  bf16* w1p = wp;                 // 64  x 32
  bf16* w2p = w1p + 64 * 32;      // 64  x 64
  bf16* w3p = w2p + 64 * 64;      // 64  x 128
  bf16* w4p = w3p + 64 * 128;     // 64  x 64
  bf16* w5p = w4p + 64 * 64;      // 64  x 128
  bf16* w6p = w5p + 64 * 128;     // 1024 x 192
  bf16* w7p = w6p + 1024 * 192;   // 512 x 192 (cols 1024..1215 of w7)
  bf16* w8p = w7p + 512 * 192;    // 256 x 512
  size_t packed_elems = 64 * 32 + 64 * 64 + 64 * 128 + 64 * 64 + 64 * 128 +
                        1024 * 192 + 512 * 192 + 256 * 512;   // 452,608
  float* ws  = (float*)(wp + packed_elems);
  int*   idx = (int*)ws;                                   // B*N*20
  float* cat = ws + (size_t)BB * NN * KNBR;                // B*N*192 (x1|x2|x3)
  float* g   = cat + (size_t)BB * NN * 192;                // B*1024
  float* gw  = g + BB * 1024;                              // B*512
  float* h6  = gw + BB * 512;                              // B*N*1024
  float* h7  = h6;                                         // reuse after g/gw
  float* h8  = h6 + (size_t)BB * NN * 512;                 // B*N*256

  dim3 blk(256);
  int totalPts = BB * NN;

  // ---- pack weights to padded bf16 ----
  auto packgrid = [](int n) { return dim3((n + 255) / 256); };
  pack_w_kernel<<<packgrid(64 * 32), blk, 0, stream>>>(w[1], 12, 0, 12, 32, 64, w1p);
  pack_w_kernel<<<packgrid(64 * 64), blk, 0, stream>>>(w[2], 64, 0, 64, 64, 64, w2p);
  pack_w_kernel<<<packgrid(64 * 128), blk, 0, stream>>>(w[3], 128, 0, 128, 128, 64, w3p);
  pack_w_kernel<<<packgrid(64 * 64), blk, 0, stream>>>(w[4], 64, 0, 64, 64, 64, w4p);
  pack_w_kernel<<<packgrid(64 * 128), blk, 0, stream>>>(w[5], 128, 0, 128, 128, 64, w5p);
  pack_w_kernel<<<packgrid(1024 * 192), blk, 0, stream>>>(w[6], 192, 0, 192, 192, 1024, w6p);
  pack_w_kernel<<<packgrid(512 * 192), blk, 0, stream>>>(w[7], 1216, 1024, 192, 192, 512, w7p);
  pack_w_kernel<<<packgrid(256 * 512), blk, 0, stream>>>(w[8], 512, 0, 512, 512, 256, w8p);

  // ---- stage 1 ----
  knn_kernel<6><<<(totalPts + 255) / 256, blk, 0, stream>>>(x, 6, 0, idx);
  edge_conv_kernel<6, true><<<BB * (NN / 16), blk, 0, stream>>>(
      x, 6, 0, idx, w1p, sc[1], bi[1], w2p, sc[2], bi[2], cat, 192, 0);
  // ---- stage 2 ----
  knn_kernel<64><<<(totalPts + 255) / 256, blk, 0, stream>>>(cat, 192, 0, idx);
  edge_conv_kernel<64, true><<<BB * (NN / 16), blk, 0, stream>>>(
      cat, 192, 0, idx, w3p, sc[3], bi[3], w4p, sc[4], bi[4], cat, 192, 64);
  // ---- stage 3 ----
  knn_kernel<64><<<(totalPts + 255) / 256, blk, 0, stream>>>(cat, 192, 64, idx);
  edge_conv_kernel<64, false><<<BB * (NN / 16), blk, 0, stream>>>(
      cat, 192, 64, idx, w5p, sc[5], bi[5], nullptr, nullptr, nullptr,
      cat, 192, 128);

  // ---- w6: cat(192) -> h6(1024), lrelu ----
  gemm_kernel<<<dim3(totalPts / 128, 1024 / 16), blk, 0, stream>>>(
      cat, 192, w6p, 192, sc[6], bi[6], nullptr, 1024, h6, 1024);
  // global max feature; fold broadcast half of w7 into per-(b,o) bias
  max_over_n_kernel<<<dim3(BB, 1024 / 256), blk, 0, stream>>>(h6, g);
  gw_kernel<<<dim3(512 / 256, BB), blk, 0, stream>>>(w[7], g, gw);
  // ---- w7: cat(192) + gw -> h7(512), lrelu ----
  gemm_kernel<<<dim3(totalPts / 128, 512 / 16), blk, 0, stream>>>(
      cat, 192, w7p, 192, sc[7], bi[7], gw, 512, h7, 512);
  // ---- w8: h7(512) -> h8(256), lrelu ----
  gemm_kernel<<<dim3(totalPts / 128, 256 / 16), blk, 0, stream>>>(
      h7, 512, w8p, 512, sc[8], bi[8], nullptr, 256, h8, 256);
  // ---- head: h8(256) -> out (B, 13, N) ----
  head_kernel<<<(BB * NN * NCLS + 255) / 256, blk, 0, stream>>>(h8, w9, b9, outp);
}
